// CausalSelfAttention_7267084665273
// MI455X (gfx1250) — matrix operand order repaired and checked by reading, not compiled, not run on hardware
//
#include <hip/hip_runtime.h>
#include <hip/hip_bf16.h>

// B=4, S=2048, E=1024, H=16, D=64
#define BB 4
#define SS 2048
#define EE 1024
#define HH 16
#define DD 64

typedef __attribute__((ext_vector_type(16))) __bf16 v16bf;
typedef __attribute__((ext_vector_type(8)))  float  v8f;
typedef __attribute__((ext_vector_type(4)))  int    v4i;

#define GLOBAL_AS __attribute__((address_space(1)))
#define LDS_AS    __attribute__((address_space(3)))

#if defined(__AMDGCN__) && __has_builtin(__builtin_amdgcn_global_load_async_to_lds_b128)
#define HAS_ASYNC 1
#else
#define HAS_ASYNC 0
#endif

// ---------- helpers ----------
// async global->LDS copy of 16 bytes per lane (ASYNCcnt path); sync fallback
__device__ __forceinline__ void cp_b128(const __bf16* g, __bf16* l) {
#if HAS_ASYNC
    __builtin_amdgcn_global_load_async_to_lds_b128(
        (GLOBAL_AS v4i*)g, (LDS_AS v4i*)l, 0, 0);
#else
    *(uint4*)l = *(const uint4*)g;
#endif
}
__device__ __forceinline__ void async_wait() {
#if HAS_ASYNC
#if __has_builtin(__builtin_amdgcn_s_wait_asynccnt)
    __builtin_amdgcn_s_wait_asynccnt(0);
#else
    asm volatile("s_wait_asynccnt 0x0" ::: "memory");
#endif
#endif
}

__device__ __forceinline__ v16bf ldfrag(const __bf16* p0, const __bf16* p1) {
    union { v16bf v; uint4 q[2]; } u;
    u.q[0] = *(const uint4*)p0;
    u.q[1] = *(const uint4*)p1;
    return u.v;
}

__device__ __forceinline__ float redmax16(float v) {
    #pragma unroll
    for (int o = 8; o >= 1; o >>= 1) v = fmaxf(v, __shfl_xor(v, o, 32));
    return v;
}
__device__ __forceinline__ float redsum16(float v) {
    #pragma unroll
    for (int o = 8; o >= 1; o >>= 1) v += __shfl_xor(v, o, 32);
    return v;
}

// ---------- f32 -> bf16 convert ----------
__global__ void cvt_f32_bf16(const float* __restrict__ in, __bf16* __restrict__ out, int n) {
    int i = (blockIdx.x * blockDim.x + threadIdx.x) * 4;
    if (i < n) {
        float4 v = *(const float4*)(in + i);
        __bf16 o[4] = { (__bf16)v.x, (__bf16)v.y, (__bf16)v.z, (__bf16)v.w };
        *(uint2*)(out + i) = *(uint2*)o;
    }
}

// ---------- bf16 GEMM: C[M,N] = A[M,K] @ B[K,N] + bias ----------
// block tile 128x64, 256 threads (8 waves), wave tile 32x32, K stage 64,
// double-buffered LDS with async global->LDS loads for A.
template<int OUTF32>
__global__ __launch_bounds__(256)
void gemm_bf16(const __bf16* __restrict__ A, const __bf16* __restrict__ B,
               const float* __restrict__ bias, void* __restrict__ Cout,
               int M, int N, int K) {
    __shared__ __align__(16) __bf16 As[2][128 * 72]; // [m][k0..63], stride 72 (144B)
    __shared__ __align__(16) __bf16 Bs[2][64 * 72];  // [n][k0..63] transposed

    const int tid  = threadIdx.x;
    const int lane = tid & 31;
    const int w    = tid >> 5;
    const int hi   = lane >> 4;
    const int l16  = lane & 15;
    const int wm   = w & 3;       // M offset 32*wm
    const int wn   = w >> 2;      // N offset 32*wn
    const int m0   = blockIdx.y * 128;
    const int n0   = blockIdx.x * 64;

    const int arow = tid >> 1, akh = (tid & 1) * 32;  // A: 32 bf16 per thread
    const int bk   = tid & 63, bn8 = (tid >> 6) * 16; // B: 16 bf16 per thread

    auto loadA = [&](int k0, int buf) {
        const __bf16* g = A + (size_t)(m0 + arow) * K + k0 + akh;
        __bf16* l = &As[buf][arow * 72 + akh];
        cp_b128(g, l); cp_b128(g + 8, l + 8);
        cp_b128(g + 16, l + 16); cp_b128(g + 24, l + 24);
    };
    auto loadB = [&](int k0, int buf) {
        const __bf16* g = B + (size_t)(k0 + bk) * N + n0 + bn8;
        uint4 u0 = ((const uint4*)g)[0], u1 = ((const uint4*)g)[1];
        __bf16 tmp[16]; *(uint4*)tmp = u0; *(uint4*)(tmp + 8) = u1;
        #pragma unroll
        for (int j = 0; j < 16; ++j) Bs[buf][(bn8 + j) * 72 + bk] = tmp[j];
    };

    v8f c[2][2] = {};
    loadA(0, 0); loadB(0, 0);
    const int nst = K / 64;
    for (int st = 0; st < nst; ++st) {
        async_wait();
        __syncthreads();                 // stage st resident block-wide
        if (st + 1 < nst) {              // prefetch next stage into other buffer
            loadA((st + 1) * 64, (st + 1) & 1);
            loadB((st + 1) * 64, (st + 1) & 1);
        }
        const __bf16* as = As[st & 1];
        const __bf16* bs = Bs[st & 1];
        #pragma unroll
        for (int s = 0; s < 2; ++s) {    // two K=32 slices
            const __bf16* a0p = as + (wm * 32 + l16) * 72 + s * 32 + hi * 8;
            const __bf16* a1p = a0p + 16 * 72;
            v16bf a0 = ldfrag(a0p, a0p + 16);
            v16bf a1 = ldfrag(a1p, a1p + 16);
            const __bf16* b0p = bs + (wn * 32 + l16) * 72 + s * 32 + hi * 8;
            const __bf16* b1p = b0p + 16 * 72;
            v16bf b0 = ldfrag(b0p, b0p + 16);
            v16bf b1 = ldfrag(b1p, b1p + 16);
            c[0][0] = __builtin_amdgcn_wmma_f32_16x16x32_bf16(false, a0, false, b0, (short)0, c[0][0], false, false);
            c[0][1] = __builtin_amdgcn_wmma_f32_16x16x32_bf16(false, a0, false, b1, (short)0, c[0][1], false, false);
            c[1][0] = __builtin_amdgcn_wmma_f32_16x16x32_bf16(false, a1, false, b0, (short)0, c[1][0], false, false);
            c[1][1] = __builtin_amdgcn_wmma_f32_16x16x32_bf16(false, a1, false, b1, (short)0, c[1][1], false, false);
        }
    }

    #pragma unroll
    for (int tm = 0; tm < 2; ++tm) {
        #pragma unroll
        for (int tn = 0; tn < 2; ++tn) {
            int col = n0 + wn * 32 + tn * 16 + l16;
            float bc = bias[col];
            #pragma unroll
            for (int r = 0; r < 8; ++r) {
                int row = m0 + wm * 32 + tm * 16 + r + hi * 8;
                float v = c[tm][tn][r] + bc;
                if (OUTF32) ((float*)Cout)[(size_t)row * N + col] = v;
                else        ((__bf16*)Cout)[(size_t)row * N + col] = (__bf16)v;
            }
        }
    }
}

// ---------- flash attention ----------
// grid: (S/64, B*H); block: 128 threads (4 waves), each wave owns 16 query rows
__global__ __launch_bounds__(128)
void flash_attn(const __bf16* __restrict__ qkvb, const int* __restrict__ amask,
                __bf16* __restrict__ attnb) {
    __shared__ __align__(16) __bf16 Qs[64 * 72];  // [qrow][d]
    __shared__ __align__(16) __bf16 Ks[32 * 72];  // [key][d]
    __shared__ __align__(16) __bf16 Vs[64 * 40];  // [d][key] transposed
    __shared__ __align__(16) __bf16 Ps[64 * 40];  // per-wave 16 rows x 32 keys
    __shared__ float biask[32];

    const int tid  = threadIdx.x;
    const int lane = tid & 31;
    const int w    = tid >> 5;
    const int hi   = lane >> 4;
    const int l16  = lane & 15;
    const int q0   = blockIdx.x * 64;
    const int b    = blockIdx.y / HH;
    const int h    = blockIdx.y % HH;
    const int bS   = b * SS;
    const int h64  = h * DD;

    // Q tile (64x64): async, 32 bf16 per thread
    {
        int row = tid >> 1, c0 = (tid & 1) * 32;
        const __bf16* g = qkvb + (size_t)(bS + q0 + row) * (3 * EE) + h64 + c0;
        __bf16* l = Qs + row * 72 + c0;
        cp_b128(g, l); cp_b128(g + 8, l + 8);
        cp_b128(g + 16, l + 16); cp_b128(g + 24, l + 24);
    }

    float m_i[8], l_i[8];
    v8f   O[4] = {};
    #pragma unroll
    for (int r = 0; r < 8; ++r) { m_i[r] = -1e30f; l_i[r] = 0.f; }

    const int nch = blockIdx.x * 2 + 2;  // key chunks of 32, up to q0+64
    for (int ch = 0; ch < nch; ++ch) {
        const int kc = ch * 32;
        __syncthreads();                 // previous chunk's reads done
        { // K tile 32x64: async
            int row = tid >> 2, c0 = (tid & 3) * 16;
            const __bf16* g = qkvb + (size_t)(bS + kc + row) * (3 * EE) + EE + h64 + c0;
            __bf16* l = Ks + row * 72 + c0;
            cp_b128(g, l); cp_b128(g + 8, l + 8);
        }
        { // V tile 32x64 -> transposed [d][key] (sync: needs transpose)
            int kk = tid & 31, d0 = (tid >> 5) * 16;
            const __bf16* g = qkvb + (size_t)(bS + kc + kk) * (3 * EE) + 2 * EE + h64 + d0;
            uint4 u0 = ((const uint4*)g)[0], u1 = ((const uint4*)g)[1];
            __bf16 tmp[16];
            *(uint4*)tmp = u0; *(uint4*)(tmp + 8) = u1;
            #pragma unroll
            for (int j = 0; j < 16; ++j) Vs[(d0 + j) * 40 + kk] = tmp[j];
        }
        if (tid < 32) {
            int key = kc + tid;
            biask[tid] = (amask[bS + key] != 0) ? 0.f : -1e30f;
        }
        async_wait();                    // covers Q (first iter) + K asyncs
        __syncthreads();

        // scores: 16 q-rows x 32 keys, K-dim = D = 64
        v8f sc[2];
        #pragma unroll
        for (int t = 0; t < 2; ++t) {
            v8f c = {};
            #pragma unroll
            for (int s = 0; s < 2; ++s) {
                const __bf16* ap = Qs + (w * 16 + l16) * 72 + s * 32 + hi * 8;
                v16bf a = ldfrag(ap, ap + 16);
                const __bf16* bp = Ks + (t * 16 + l16) * 72 + s * 32 + hi * 8;
                v16bf bb = ldfrag(bp, bp + 16);
                c = __builtin_amdgcn_wmma_f32_16x16x32_bf16(false, a, false, bb, (short)0, c, false, false);
            }
            float bk  = biask[t * 16 + l16];
            int   key = kc + t * 16 + l16;
            #pragma unroll
            for (int r = 0; r < 8; ++r) {
                int rowq = q0 + w * 16 + r + hi * 8;
                float v = c[r] * 0.125f + bk;       // 1/sqrt(64)
                c[r] = (key > rowq) ? -1e30f : v;   // causal
            }
            sc[t] = c;
        }

        // online softmax (row stats live in the 16-lane groups of the C layout)
        #pragma unroll
        for (int r = 0; r < 8; ++r) {
            float mx = redmax16(fmaxf(sc[0][r], sc[1][r]));
            float mnew = fmaxf(m_i[r], mx);
            float corr = __expf(m_i[r] - mnew);
            m_i[r] = mnew;
            float p0 = __expf(sc[0][r] - mnew);
            float p1 = __expf(sc[1][r] - mnew);
            float rs = redsum16(p0 + p1);
            l_i[r] = l_i[r] * corr + rs;
            #pragma unroll
            for (int t2 = 0; t2 < 4; ++t2) O[t2][r] *= corr;
            __bf16* pr = Ps + (w * 16 + r + hi * 8) * 40;
            pr[l16]      = (__bf16)p0;
            pr[16 + l16] = (__bf16)p1;
        }
        __syncthreads();

        // O += P(16x32) @ V(32x64)
        const __bf16* app = Ps + (w * 16 + l16) * 40 + hi * 8;
        v16bf ap = ldfrag(app, app + 16);
        #pragma unroll
        for (int t = 0; t < 4; ++t) {
            const __bf16* bp = Vs + (t * 16 + l16) * 40 + hi * 8;
            v16bf bv = ldfrag(bp, bp + 16);
            O[t] = __builtin_amdgcn_wmma_f32_16x16x32_bf16(false, ap, false, bv, (short)0, O[t], false, false);
        }
    }

    // normalize + store bf16 (input to projection GEMM)
    #pragma unroll
    for (int r = 0; r < 8; ++r) {
        float inv = 1.f / l_i[r];
        int row = bS + q0 + w * 16 + r + hi * 8;
        #pragma unroll
        for (int t = 0; t < 4; ++t) {
            attnb[(size_t)row * EE + h64 + t * 16 + l16] = (__bf16)(O[t][r] * inv);
        }
    }
}

// ---------- launch ----------
extern "C" void kernel_launch(void* const* d_in, const int* in_sizes, int n_in,
                              void* d_out, int out_size, void* d_ws, size_t ws_size,
                              hipStream_t stream) {
    const float* x     = (const float*)d_in[0];
    const int*   amask = (const int*)d_in[1];
    const float* Wqkv  = (const float*)d_in[2];
    const float* bqkv  = (const float*)d_in[3];
    const float* Wproj = (const float*)d_in[4];
    const float* bproj = (const float*)d_in[5];
    float*       out   = (float*)d_out;

    const size_t Mrows = (size_t)BB * SS;           // 8192
    char* wp = (char*)d_ws;
    __bf16* xb     = (__bf16*)wp; wp += Mrows * EE * 2;           // 16 MB
    __bf16* wqkvb  = (__bf16*)wp; wp += (size_t)EE * 3 * EE * 2;  // 6 MB
    __bf16* wprojb = (__bf16*)wp; wp += (size_t)EE * EE * 2;      // 2 MB
    __bf16* qkvb   = (__bf16*)wp; wp += Mrows * 3 * EE * 2;       // 48 MB
    __bf16* attnb  = (__bf16*)wp;                                 // 16 MB

    {   // fp32 -> bf16 conversions
        int n1 = (int)(Mrows * EE);
        cvt_f32_bf16<<<n1 / 1024, 256, 0, stream>>>(x, xb, n1);
        int n2 = EE * 3 * EE;
        cvt_f32_bf16<<<n2 / 1024, 256, 0, stream>>>(Wqkv, wqkvb, n2);
        int n3 = EE * EE;
        cvt_f32_bf16<<<n3 / 1024, 256, 0, stream>>>(Wproj, wprojb, n3);
    }

    // QKV projection: [8192,1024] @ [1024,3072] -> bf16
    gemm_bf16<0><<<dim3(3 * EE / 64, Mrows / 128), 256, 0, stream>>>(
        xb, wqkvb, bqkv, qkvb, (int)Mrows, 3 * EE, EE);

    // causal flash attention
    flash_attn<<<dim3(SS / 64, BB * HH), 128, 0, stream>>>(qkvb, amask, attnb);

    // output projection: [8192,1024] @ [1024,1024] -> f32 into d_out
    gemm_bf16<1><<<dim3(EE / 64, Mrows / 128), 256, 0, stream>>>(
        attnb, wprojb, bproj, out, (int)Mrows, EE, EE);
}
